// Attention_1065151889535
// MI455X (gfx1250) — matrix operand order repaired and checked by reading, not compiled, not run on hardware
//
#include <hip/hip_runtime.h>

typedef __attribute__((ext_vector_type(16))) __bf16 v16bf;
typedef __attribute__((ext_vector_type(8)))  __bf16 v8bf;
typedef __attribute__((ext_vector_type(8)))  float  v8f;
typedef __attribute__((ext_vector_type(4)))  float  v4f;

#define NRES 256
#define CQn  128
#define Hn   4
#define CHn  32
#define En   128          // Hn*CHn
#define MTOT 65536        // NRES*NRES

// ---------------- workspace layout (in __bf16 elements) ----------------
static constexpr size_t WSO_W  = 0;                                // 5 weight mats, bf16
static constexpr size_t WSO_QP = 81920;                            // q  [n,h,q,d]
static constexpr size_t SZ_PROJ = (size_t)NRES * Hn * NRES * CHn;  // 8388608
static constexpr size_t WSO_KP = WSO_QP + SZ_PROJ;                 // k  [n,h,k,d]
static constexpr size_t WSO_VT = WSO_KP + SZ_PROJ;                 // v  [n,h,d,k] (transposed)
static constexpr size_t WSO_GS = WSO_VT + SZ_PROJ;                 // sigmoid gate [m,e]
static constexpr size_t WSO_OP = WSO_GS + SZ_PROJ;                 // attn out (pre-gate) [m,e]

// ---------------- WMMA operand loaders (CDNA5 wave32 layouts) ----------------
// A (16x32, 16-bit): lanes 0-15: M=lane, K = {0..7, 16..23}; lanes 16-31: M=lane-16, K = {8..15, 24..31}
__device__ __forceinline__ v16bf load_a_bf16(const __bf16* base, int ld, int lane) {
  int r = lane & 15, hi = lane >> 4;
  const __bf16* p = base + (size_t)r * ld + hi * 8;
  v8bf x0 = *(const v8bf*)(p);
  v8bf x1 = *(const v8bf*)(p + 16);
  v16bf a;
#pragma unroll
  for (int j = 0; j < 8; ++j) { a[j] = x0[j]; a[8 + j] = x1[j]; }
  return a;
}

// A from f32 memory, converted to bf16
__device__ __forceinline__ v16bf load_a_f32(const float* base, int ld, int lane) {
  int r = lane & 15, hi = lane >> 4;
  const float* p = base + (size_t)r * ld + hi * 8;
  v4f x0 = *(const v4f*)(p);
  v4f x1 = *(const v4f*)(p + 4);
  v4f x2 = *(const v4f*)(p + 16);
  v4f x3 = *(const v4f*)(p + 20);
  v16bf a;
#pragma unroll
  for (int j = 0; j < 4; ++j) {
    a[j]      = (__bf16)x0[j];
    a[4 + j]  = (__bf16)x1[j];
    a[8 + j]  = (__bf16)x2[j];
    a[12 + j] = (__bf16)x3[j];
  }
  return a;
}

// B (32x16, 16-bit), sourced from row-major B^T (i.e. [N=16 rows][K=32]):
// lanes 0-15: N=lane, K = {0..7, 16..23} ; lanes 16-31: N=lane-16, K = {8..15, 24..31}
__device__ __forceinline__ v16bf load_b_bf16(const __bf16* baseT, int ld, int lane) {
  int r = lane & 15, hi = lane >> 4;
  const __bf16* p = baseT + (size_t)r * ld + hi * 8;
  v8bf x0 = *(const v8bf*)(p);
  v8bf x1 = *(const v8bf*)(p + 16);
  v16bf b;
#pragma unroll
  for (int j = 0; j < 8; ++j) { b[j] = x0[j]; b[8 + j] = x1[j]; }
  return b;
}

__device__ __forceinline__ v8f wmma_bf16(v16bf a, v16bf b, v8f c) {
  return __builtin_amdgcn_wmma_f32_16x16x32_bf16(false, a, false, b, (short)0, c, false, false);
}

// ---------------- kernel 0: weights f32 -> bf16 ----------------
__global__ __launch_bounds__(256) void prep_weights(
    const float* __restrict__ wq, const float* __restrict__ wk,
    const float* __restrict__ wv, const float* __restrict__ wg,
    const float* __restrict__ wo, __bf16* __restrict__ dst) {
  int i = blockIdx.x * 256 + threadIdx.x;
  if (i >= 5 * 16384) return;
  int which = i >> 14, j = i & 16383;
  const float* src = (which == 0) ? wq : (which == 1) ? wk : (which == 2) ? wv
                   : (which == 3) ? wg : wo;
  dst[i] = (__bf16)src[j];
}

// ---------------- kernel 1: q/k/v/g projections ----------------
// proj decoded from blockIdx only (SGPR -> scalar branches in the epilogue).
// One wave per 16-row strip: the 4 A K-tiles are loaded once into 32 VGPRs,
// then each of 8 col-tiles runs a 4-WMMA chain with one live accumulator.
// grid = 4096 blocks * 4 waves; proj = blockIdx.x>>10, strip = (blockIdx.x&1023)*4+wave
__global__ __launch_bounds__(128, 1) void proj_kernel(
    const float* __restrict__ q_x, const float* __restrict__ kv_x,
    const float* __restrict__ bg, __bf16* __restrict__ ws) {
  int lane = threadIdx.x & 31;
  int wave = threadIdx.x >> 5;
  int proj = blockIdx.x >> 10;                  // wave-uniform, scalar
  int mstrip = (blockIdx.x & 1023) * 4 + wave;  // 0..4095
  int m0 = mstrip << 4;

  const float* x = (proj == 0 || proj == 3) ? q_x : kv_x;
  const __bf16* wb = ws + WSO_W + (size_t)proj * 16384;

  int r = lane & 15, hi = lane >> 4;

  // A strip: 16 rows x 128 cols, held in registers as 4 K-tiles
  v16bf a[4];
#pragma unroll
  for (int kt = 0; kt < 4; ++kt)
    a[kt] = load_a_f32(x + (size_t)m0 * CQn + kt * 32, CQn, lane);

  const float scale = 0.17677669529663689f;   // 1/sqrt(32)
  __bf16* qp = ws + WSO_QP;
  __bf16* kp = ws + WSO_KP;
  __bf16* vt = ws + WSO_VT;
  __bf16* gs = ws + WSO_GS;

#pragma unroll
  for (int ct = 0; ct < 8; ++ct) {
    v8f acc = {};
#pragma unroll
    for (int kt = 0; kt < 4; ++kt) {
      v16bf b = load_b_bf16(wb + (size_t)(ct * 16) * CQn + kt * 32, CQn, lane);
      acc = wmma_bf16(a[kt], b, acc);
    }

    int e0 = ct * 16;
    int h = e0 >> 5;            // head (tile stays within one head since 16|32)
    int dcol = (e0 & 31) + r;
    int e = e0 + r;

    if (proj == 0) {            // q, scaled
#pragma unroll
      for (int v = 0; v < 8; ++v) {
        int m = m0 + v + hi * 8;
        int n = m >> 8, qr = m & 255;
        qp[(((size_t)(n * Hn + h)) * NRES + qr) * CHn + dcol] = (__bf16)(acc[v] * scale);
      }
    } else if (proj == 1) {     // k
#pragma unroll
      for (int v = 0; v < 8; ++v) {
        int m = m0 + v + hi * 8;
        int n = m >> 8, kr = m & 255;
        kp[(((size_t)(n * Hn + h)) * NRES + kr) * CHn + dcol] = (__bf16)acc[v];
      }
    } else if (proj == 2) {     // v, stored transposed [n,h,d,k]
#pragma unroll
      for (int v = 0; v < 8; ++v) {
        int m = m0 + v + hi * 8;
        int n = m >> 8, kr = m & 255;
        vt[(((size_t)(n * Hn + h)) * CHn + dcol) * NRES + kr] = (__bf16)acc[v];
      }
    } else {                    // g: sigmoid(acc + bg[e]), stored [m,e]
      float bge = bg[e];
#pragma unroll
      for (int v = 0; v < 8; ++v) {
        int m = m0 + v + hi * 8;
        float val = acc[v] + bge;
        float sig = 1.0f / (1.0f + __expf(-val));
        gs[(size_t)m * En + e] = (__bf16)sig;
      }
    }
  }
}

// ---------------- kernel 2: attention core ----------------
// grid.x = 8192 = (n*4+h)*8 + qblock(32 rows); 64 threads = 2 waves, 16 q-rows each.
__global__ __launch_bounds__(64, 1) void attn_kernel(
    const float* __restrict__ mask_bias,
    const float* __restrict__ tri_bias,
    __bf16* __restrict__ ws) {
  __shared__ float  Sbuf[2][16 * 256];
  __shared__ __bf16 Pbuf[2][16 * 256];
  __shared__ float  rstat[2][16];

  int lane = threadIdx.x & 31;
  int wave = threadIdx.x >> 5;
  int qb = blockIdx.x & 7;
  int nh = blockIdx.x >> 3;
  int n = nh >> 2, h = nh & 3;
  int q0 = qb * 32 + wave * 16;

  const __bf16* qp = ws + WSO_QP + (((size_t)nh) * NRES + q0) * CHn;
  const __bf16* kp = ws + WSO_KP + ((size_t)nh) * NRES * CHn;
  const __bf16* vt = ws + WSO_VT + ((size_t)nh) * CHn * NRES;

  int r = lane & 15, hi = lane >> 4;

  // warm the V^T panel into cache while the score WMMAs run
  __builtin_prefetch(vt + (size_t)lane * NRES, 0, 3);
  __builtin_prefetch(vt + (size_t)lane * NRES + 128, 0, 3);

  // ---- scores: S[16 x 256] = Q(16x32) * K^T, + mask + triangle ----
  v16bf a_q = load_a_bf16(qp, CHn, lane);
#pragma unroll 4
  for (int kt = 0; kt < 16; ++kt) {
    v16bf b_k = load_b_bf16(kp + (size_t)(kt * 16) * CHn, CHn, lane);
    v8f s = {};
    s = wmma_bf16(a_q, b_k, s);
    int kcol = kt * 16 + r;
    float mb = mask_bias[(size_t)n * NRES + kcol];
    const float* trow = tri_bias + ((size_t)h * NRES) * NRES + kcol;
#pragma unroll
    for (int v = 0; v < 8; ++v) {
      int qloc = v + hi * 8;
      int qg = q0 + qloc;
      float val = s[v] + mb + trow[(size_t)qg * NRES];
      Sbuf[wave][qloc * 256 + kcol] = val;
    }
  }

  // ---- softmax over k (two half-rows per row, combined with shfl_xor 16) ----
  {
    const float* row = &Sbuf[wave][r * 256 + hi * 128];
    float mx = -3.0e38f;
#pragma unroll 8
    for (int j = 0; j < 128; ++j) mx = fmaxf(mx, row[j]);
    mx = fmaxf(mx, __shfl_xor(mx, 16, 32));
    float sum = 0.0f;
    __bf16* prow = &Pbuf[wave][r * 256 + hi * 128];
#pragma unroll 8
    for (int j = 0; j < 128; ++j) {
      float p = __expf(row[j] - mx);
      sum += p;
      prow[j] = (__bf16)p;
    }
    sum += __shfl_xor(sum, 16, 32);
    if (hi == 0) rstat[wave][r] = 1.0f / sum;
  }

  // ---- O = P(16x256) * V(256x32), V supplied transposed ----
  v8f o0 = {}, o1 = {};
#pragma unroll 4
  for (int kt = 0; kt < 8; ++kt) {
    v16bf a_p = load_a_bf16(&Pbuf[wave][kt * 32], 256, lane);
    v16bf b0 = load_b_bf16(vt + (size_t)0 * NRES + kt * 32, NRES, lane);
    v16bf b1 = load_b_bf16(vt + (size_t)16 * NRES + kt * 32, NRES, lane);
    o0 = wmma_bf16(a_p, b0, o0);
    o1 = wmma_bf16(a_p, b1, o1);
  }

  __bf16* op = ws + WSO_OP;
#pragma unroll
  for (int v = 0; v < 8; ++v) {
    int qloc = v + hi * 8;
    float rs = rstat[wave][qloc];
    size_t base = ((size_t)n * NRES + (q0 + qloc)) * En + h * CHn;
    op[base + r]      = (__bf16)(o0[v] * rs);
    op[base + 16 + r] = (__bf16)(o1[v] * rs);
  }
}

// ---------------- kernel 3: gated output projection ----------------
// out[m,c] = sum_e (o[m,e]*g[m,e]) * wo[c,e] + bo[c]
// One wave per 16-row strip: gated A tiles computed once (32 VGPRs), then one
// accumulator per col-tile with immediate store. grid = 1024 blocks * 4 waves.
__global__ __launch_bounds__(128, 1) void out_kernel(
    const float* __restrict__ bo, const __bf16* __restrict__ ws,
    float* __restrict__ out) {
  int lane = threadIdx.x & 31;
  int w = blockIdx.x * 4 + (threadIdx.x >> 5);
  int m0 = w << 4;

  const __bf16* op = ws + WSO_OP;
  const __bf16* gs = ws + WSO_GS;
  const __bf16* wob = ws + WSO_W + 4 * (size_t)16384;

  int r = lane & 15, hi = lane >> 4;

  // gated A strip (o * g), held in registers as 4 K-tiles
  v16bf a[4];
#pragma unroll
  for (int kt = 0; kt < 4; ++kt) {
    v16bf ao = load_a_bf16(op + (size_t)m0 * En + kt * 32, En, lane);
    v16bf ag = load_a_bf16(gs + (size_t)m0 * En + kt * 32, En, lane);
    v16bf p;
#pragma unroll
    for (int j = 0; j < 16; ++j) p[j] = (__bf16)((float)ao[j] * (float)ag[j]);
    a[kt] = p;
  }

#pragma unroll
  for (int ct = 0; ct < 8; ++ct) {
    v8f acc = {};
#pragma unroll
    for (int kt = 0; kt < 4; ++kt) {
      v16bf b = load_b_bf16(wob + (size_t)(ct * 16) * En + kt * 32, En, lane);
      acc = wmma_bf16(a[kt], b, acc);
    }
    int c0 = ct * 16;
    float bias = bo[c0 + r];
#pragma unroll
    for (int v = 0; v < 8; ++v) {
      int m = m0 + v + hi * 8;
      out[(size_t)m * CQn + (c0 + r)] = acc[v] + bias;
    }
  }
}

// ---------------- host launcher ----------------
extern "C" void kernel_launch(void* const* d_in, const int* in_sizes, int n_in,
                              void* d_out, int out_size, void* d_ws, size_t ws_size,
                              hipStream_t stream) {
  const float* q_x      = (const float*)d_in[0];
  const float* kv_x     = (const float*)d_in[1];
  const float* mask     = (const float*)d_in[2];
  const float* tri      = (const float*)d_in[3];
  const float* wq       = (const float*)d_in[4];
  const float* wk       = (const float*)d_in[5];
  const float* wv       = (const float*)d_in[6];
  const float* wg       = (const float*)d_in[7];
  const float* bg       = (const float*)d_in[8];
  const float* wo       = (const float*)d_in[9];
  const float* bo       = (const float*)d_in[10];
  float* out = (float*)d_out;
  __bf16* ws = (__bf16*)d_ws;

  prep_weights<<<320, 256, 0, stream>>>(wq, wk, wv, wg, wo, ws);
  proj_kernel<<<4096, 128, 0, stream>>>(q_x, kv_x, bg, ws);
  attn_kernel<<<8192, 64, 0, stream>>>(mask, tri, ws);
  out_kernel<<<1024, 128, 0, stream>>>(bo, ws, out);
}
